// LSTM_55989193670929
// MI455X (gfx1250) — matrix operand order repaired and checked by reading, not compiled, not run on hardware
//
#include <hip/hip_runtime.h>

// MI455X (gfx1250) LSTM: f16 WMMA GEMMs with f32 accumulation.
// Phase 1: pack inputs/W/U into WMMA-fragment-ordered f16 buffers.
// Phase 2: x_proj = inputs @ [Wi|Wf|Wo|Wc] + b   (big GEMM, v_wmma_f32_16x16x32_f16)
// Phase 3: ONE persistent kernel runs all 512 recurrence steps: U fragments and
//          c-state live in registers across steps; grid-wide atomic barrier
//          between steps; h ping-pongs through two L2-hot 64KB buffers.

typedef __attribute__((ext_vector_type(16))) _Float16 v16h;
typedef __attribute__((ext_vector_type(8)))  float    v8f;

namespace {
constexpr int T_SEQ = 512;
constexpr int B_SZ  = 64;
constexpr int I_SZ  = 512;
constexpr int H_SZ  = 512;
constexpr int G4    = 4 * H_SZ;       // 2048 gate columns
constexpr int KT    = I_SZ / 32;      // 16 K-tiles (I == H)
constexpr int MROWS = B_SZ * T_SEQ;   // 32768 rows of the input GEMM
constexpr int NBLK  = 16;             // persistent scan blocks
}

// ---- WMMA fragment address helpers (per CDNA5 ISA §7.12.2) ----
__device__ __forceinline__ int a_frag_pos(int m, int k) {
  int hi = ((k & 15) >= 8) ? 1 : 0;
  int k0 = k - 8 * hi;                       // {0..7, 16..23}
  int j  = (k0 < 8) ? (k0 >> 1) : (4 + ((k0 - 16) >> 1));
  int e  = 2 * j + (k0 & 1);
  return (m + 16 * hi) * 16 + e;             // lane*16 + element
}
__device__ __forceinline__ int b_frag_pos(int n, int k) {
  return (n + 16 * ((k >> 3) & 1)) * 16 + ((k >> 4) << 3) + (k & 7);
}

__device__ __forceinline__ float sigm(float x) { return 1.0f / (1.0f + __expf(-x)); }
__device__ __forceinline__ float tanh_f(float x) {
  x = fminf(fmaxf(x, -15.0f), 15.0f);
  float e = __expf(2.0f * x);
  return (e - 1.0f) / (e + 1.0f);
}

// ---- Pack inputs [B,T,I] f32 -> A-fragment-ordered f16 ----
__global__ void __launch_bounds__(256)
pack_inputs(const float* __restrict__ x, _Float16* __restrict__ af) {
  int id = blockIdx.x * 256 + threadIdx.x;       // < MROWS * I_SZ
  int r = id >> 9;                               // row = b*T + t
  int k = id & 511;
  float v = x[id];
  int mt = r >> 4, kt = k >> 5;
  int pos = a_frag_pos(r & 15, k & 31);
  af[(size_t)(mt * KT + kt) * 512 + pos] = (_Float16)v;
}

// ---- Pack [512, 2048] concat of 4 weight mats -> B-fragment-ordered f16 ----
__global__ void __launch_bounds__(256)
pack_wu(const float* __restrict__ W0, const float* __restrict__ W1,
        const float* __restrict__ W2, const float* __restrict__ W3,
        const float* __restrict__ b0, const float* __restrict__ b1,
        const float* __restrict__ b2, const float* __restrict__ b3,
        _Float16* __restrict__ bfrag, float* __restrict__ bias) {
  int id = blockIdx.x * 256 + threadIdx.x;       // < 512 * 2048
  int kg = id >> 11;                             // K row 0..511
  int ng = id & 2047;                            // gate column 0..2047
  const float* W = (ng < 512) ? W0 : (ng < 1024) ? W1 : (ng < 1536) ? W2 : W3;
  float v = W[kg * 512 + (ng & 511)];
  int nt = ng >> 4, kt = kg >> 5;
  int pos = b_frag_pos(ng & 15, kg & 31);
  bfrag[(size_t)(nt * KT + kt) * 512 + pos] = (_Float16)v;
  if (bias && id < G4) {
    const float* bs = (id < 512) ? b0 : (id < 1024) ? b1 : (id < 1536) ? b2 : b3;
    bias[id] = bs[id & 511];
  }
}

__global__ void __launch_bounds__(256)
init_state(_Float16* __restrict__ h0, _Float16* __restrict__ h1,
           unsigned* __restrict__ bar) {
  int id = blockIdx.x * 256 + threadIdx.x;       // < B*H
  h0[id] = (_Float16)0.0f;
  h1[id] = (_Float16)0.0f;
  if (id == 0) *bar = 0u;
}

// ---- Big GEMM: x_proj[t][b][g] = sum_k inputs[r][k] * W[k][g] + bias[g] ----
__global__ void __launch_bounds__(256)
xproj_gemm(const _Float16* __restrict__ af, const _Float16* __restrict__ wfrag,
           const float* __restrict__ bias, _Float16* __restrict__ xproj) {
  const int lane = threadIdx.x & 31;
  const int wave = threadIdx.x >> 5;
  const int mt  = blockIdx.y * 8 + wave;         // 0..2047
  const int nt0 = blockIdx.x * 4;                // 0..124 step 4

  const v16h* A  = (const v16h*)af + (size_t)(mt * KT) * 32 + lane;
  const v16h* B0 = (const v16h*)wfrag + (size_t)((nt0 + 0) * KT) * 32 + lane;
  const v16h* B1 = (const v16h*)wfrag + (size_t)((nt0 + 1) * KT) * 32 + lane;
  const v16h* B2 = (const v16h*)wfrag + (size_t)((nt0 + 2) * KT) * 32 + lane;
  const v16h* B3 = (const v16h*)wfrag + (size_t)((nt0 + 3) * KT) * 32 + lane;

  v8f acc0 = {0,0,0,0,0,0,0,0}, acc1 = acc0, acc2 = acc0, acc3 = acc0;

  for (int kt = 0; kt < KT; ++kt) {
    __builtin_prefetch((const void*)(A + (kt + 2) * 32), 0, 1);
    v16h a  = A[kt * 32];
    v16h b0 = B0[kt * 32];
    v16h b1 = B1[kt * 32];
    v16h b2 = B2[kt * 32];
    v16h b3 = B3[kt * 32];
    acc0 = __builtin_amdgcn_wmma_f32_16x16x32_f16(false, a, false, b0, (short)0, acc0, false, false);
    acc1 = __builtin_amdgcn_wmma_f32_16x16x32_f16(false, a, false, b1, (short)0, acc1, false, false);
    acc2 = __builtin_amdgcn_wmma_f32_16x16x32_f16(false, a, false, b2, (short)0, acc2, false, false);
    acc3 = __builtin_amdgcn_wmma_f32_16x16x32_f16(false, a, false, b3, (short)0, acc3, false, false);
  }

  const int hi = lane >> 4;
  const int l  = lane & 15;
  const int c0 = (nt0 + 0) * 16 + l, c1 = (nt0 + 1) * 16 + l;
  const int c2 = (nt0 + 2) * 16 + l, c3 = (nt0 + 3) * 16 + l;
  const float bb0 = bias[c0], bb1 = bias[c1], bb2 = bias[c2], bb3 = bias[c3];
  const int rbase = mt * 16 + 8 * hi;
#pragma unroll
  for (int v = 0; v < 8; ++v) {
    int r = rbase + v;
    int bi = r >> 9;                             // batch
    int t  = r & 511;                            // time
    size_t ro = ((size_t)t * B_SZ + bi) * G4;
    xproj[ro + c0] = (_Float16)(acc0[v] + bb0);
    xproj[ro + c1] = (_Float16)(acc1[v] + bb1);
    xproj[ro + c2] = (_Float16)(acc2[v] + bb2);
    xproj[ro + c3] = (_Float16)(acc3[v] + bb3);
  }
}

// ---- Persistent scan: all 512 steps in one kernel, 16 resident blocks ----
// Block jb owns hidden columns [jb*32, jb*32+32) for all 4 gates.
// Wave w: gate s = w>>1, 16-col half hs = w&1 -> one N-tile, 4 M-tiles.
__global__ void __launch_bounds__(256)
lstm_scan(_Float16* __restrict__ h0buf, _Float16* __restrict__ h1buf,
          const _Float16* __restrict__ ufrag, const _Float16* __restrict__ xproj,
          float* __restrict__ hfinal, unsigned* __restrict__ bar) {
  const int lane = threadIdx.x & 31;
  const int wave = threadIdx.x >> 5;
  const int jb = blockIdx.x;                     // 0..15
  const int s  = wave >> 1;                      // gate 0..3
  const int hs = wave & 1;                       // 16-col half
  const int nt = s * 32 + jb * 2 + hs;           // global N-tile

  __shared__ _Float16 gbuf[64 * 128];            // 16 KB gate staging

  // U fragments for this wave's N-tile, all 16 K-tiles: resident in VGPRs
  // for the entire 512-step scan (~128 VGPRs).
  const v16h* B = (const v16h*)ufrag;
  v16h breg[KT];
#pragma unroll
  for (int kt = 0; kt < KT; ++kt)
    breg[kt] = B[(size_t)(nt * KT + kt) * 32 + lane];

  // c-state lives in registers for the whole scan (8 elements/thread).
  float creg[8];
#pragma unroll
  for (int u = 0; u < 8; ++u) creg[u] = 0.0f;

  const int hi = lane >> 4;
  const int l  = lane & 15;
  const int col = s * 32 + hs * 16 + l;          // block-local gbuf column

  for (int t = 0; t < T_SEQ; ++t) {
    const _Float16* hin = (t & 1) ? h1buf : h0buf;
    _Float16*       hout = (t & 1) ? h0buf : h1buf;
    const v16h* A = (const v16h*)hin;

    // Prefetch next step's x_proj slice for this block while we compute.
    {
      int tn = (t + 1 < T_SEQ) ? (t + 1) : t;
      int pb = threadIdx.x >> 2, ps = threadIdx.x & 3;
      __builtin_prefetch(
          (const void*)(xproj + ((size_t)tn * B_SZ + pb) * G4 + ps * H_SZ + jb * 32), 0, 1);
    }

    v8f acc0 = {0,0,0,0,0,0,0,0}, acc1 = acc0, acc2 = acc0, acc3 = acc0;
#pragma unroll
    for (int kt = 0; kt < KT; ++kt) {
      v16h a0 = A[(size_t)(0 * KT + kt) * 32 + lane];
      v16h a1 = A[(size_t)(1 * KT + kt) * 32 + lane];
      v16h a2 = A[(size_t)(2 * KT + kt) * 32 + lane];
      v16h a3 = A[(size_t)(3 * KT + kt) * 32 + lane];
      acc0 = __builtin_amdgcn_wmma_f32_16x16x32_f16(false, a0, false, breg[kt], (short)0, acc0, false, false);
      acc1 = __builtin_amdgcn_wmma_f32_16x16x32_f16(false, a1, false, breg[kt], (short)0, acc1, false, false);
      acc2 = __builtin_amdgcn_wmma_f32_16x16x32_f16(false, a2, false, breg[kt], (short)0, acc2, false, false);
      acc3 = __builtin_amdgcn_wmma_f32_16x16x32_f16(false, a3, false, breg[kt], (short)0, acc3, false, false);
    }

    // Spill gate pre-activations to LDS (block-local [64 x 128]).
#pragma unroll
    for (int v = 0; v < 8; ++v) {
      gbuf[( 0 + v + 8 * hi) * 128 + col] = (_Float16)acc0[v];
      gbuf[(16 + v + 8 * hi) * 128 + col] = (_Float16)acc1[v];
      gbuf[(32 + v + 8 * hi) * 128 + col] = (_Float16)acc2[v];
      gbuf[(48 + v + 8 * hi) * 128 + col] = (_Float16)acc3[v];
    }
    __syncthreads();

    // Gate math + state update: 2048 (b, jl) elements, 8 per thread.
#pragma unroll
    for (int u = 0; u < 8; ++u) {
      int idx = u * 256 + threadIdx.x;
      int b  = idx >> 5;
      int jl = idx & 31;
      size_t xb = ((size_t)t * B_SZ + b) * G4 + jb * 32 + jl;
      float g0 = (float)gbuf[b * 128 +  0 + jl] + (float)xproj[xb + 0 * H_SZ];
      float g1 = (float)gbuf[b * 128 + 32 + jl] + (float)xproj[xb + 1 * H_SZ];
      float g2 = (float)gbuf[b * 128 + 64 + jl] + (float)xproj[xb + 2 * H_SZ];
      float g3 = (float)gbuf[b * 128 + 96 + jl] + (float)xproj[xb + 3 * H_SZ];
      float ig = sigm(g0), fg = sigm(g1), og = sigm(g2);
      float ct = tanh_f(g3);
      float cn = fg * creg[u] + ig * ct;
      creg[u] = cn;
      float h = og * tanh_f(cn);
      int jg = jb * 32 + jl;
      int mt2 = b >> 4, kt2 = jg >> 5;
      hout[(size_t)(mt2 * KT + kt2) * 512 + a_frag_pos(b & 15, jg & 31)] = (_Float16)h;
      if (t == T_SEQ - 1) hfinal[b * H_SZ + jg] = h;
    }

    // Grid-wide barrier: release h writes, arrive, spin, acquire.
    __threadfence();
    __syncthreads();
    if (threadIdx.x == 0) {
      atomicAdd(bar, 1u);
      unsigned expected = (unsigned)(t + 1) * (unsigned)NBLK;
      while (__hip_atomic_load(bar, __ATOMIC_RELAXED, __HIP_MEMORY_SCOPE_AGENT) < expected)
        __builtin_amdgcn_s_sleep(1);
    }
    __syncthreads();
    __threadfence();
  }
}

extern "C" void kernel_launch(void* const* d_in, const int* in_sizes, int n_in,
                              void* d_out, int out_size, void* d_ws, size_t ws_size,
                              hipStream_t stream) {
  const float* inp = (const float*)d_in[0];
  const float* Wi = (const float*)d_in[1], *Wf = (const float*)d_in[2];
  const float* Wo = (const float*)d_in[3], *Wc = (const float*)d_in[4];
  const float* Ui = (const float*)d_in[5], *Uf = (const float*)d_in[6];
  const float* Uo = (const float*)d_in[7], *Uc = (const float*)d_in[8];
  const float* bi = (const float*)d_in[9], *bf = (const float*)d_in[10];
  const float* bo = (const float*)d_in[11], *bc = (const float*)d_in[12];

  char* ws = (char*)d_ws;
  size_t off = 0;
  auto carve = [&](size_t bytes) -> char* {
    char* p = ws + off;
    off = (off + bytes + 255) & ~(size_t)255;
    return p;
  };
  _Float16* af    = (_Float16*)carve((size_t)MROWS * I_SZ * 2);      //  32 MB
  _Float16* wfrag = (_Float16*)carve((size_t)I_SZ * G4 * 2);         //   2 MB
  _Float16* ufrag = (_Float16*)carve((size_t)H_SZ * G4 * 2);         //   2 MB
  float*    bias  = (float*)   carve((size_t)G4 * 4);                //   8 KB
  _Float16* xproj = (_Float16*)carve((size_t)T_SEQ * B_SZ * G4 * 2); // 128 MB
  _Float16* h0    = (_Float16*)carve((size_t)B_SZ * H_SZ * 2);       //  64 KB
  _Float16* h1    = (_Float16*)carve((size_t)B_SZ * H_SZ * 2);       //  64 KB
  unsigned* bar   = (unsigned*)carve(256);
  (void)ws_size; (void)in_sizes; (void)n_in; (void)out_size;

  pack_inputs<<<(MROWS * I_SZ) / 256, 256, 0, stream>>>(inp, af);
  pack_wu<<<(I_SZ * G4) / 256, 256, 0, stream>>>(Wi, Wf, Wo, Wc, bi, bf, bo, bc, wfrag, bias);
  pack_wu<<<(H_SZ * G4) / 256, 256, 0, stream>>>(Ui, Uf, Uo, Uc, nullptr, nullptr, nullptr, nullptr, ufrag, nullptr);
  init_state<<<(B_SZ * H_SZ) / 256, 256, 0, stream>>>(h0, h1, bar);

  dim3 g1(G4 / 64, MROWS / 128);
  xproj_gemm<<<g1, 256, 0, stream>>>(af, wfrag, bias, xproj);

  lstm_scan<<<NBLK, 256, 0, stream>>>(h0, h1, ufrag, xproj, (float*)d_out, bar);
}
